// GPRGNN_13683765805696
// MI455X (gfx1250) — matrix operand order repaired and checked by reading, not compiled, not run on hardware
//
#include <hip/hip_runtime.h>
#include <math.h>

#define ALPHA 0.1f
#define KPROP 10
#define INCH  500
#define HID   256
#define OUTCH 64
#define KPAD  512   // 500 rounded up to multiple of 32
#define NKT   (KPAD / 32)

typedef __attribute__((ext_vector_type(16))) __bf16 v16bf;
typedef __attribute__((ext_vector_type(8)))  float  v8f;

union FragU { uint4 q[2]; v16bf v; };

// A fragment (bf16 16x32, row-major source [rows][stride]):
// lane l<16 : row m=l, VGPRs 0-3 = K 0..7, VGPRs 4-7 = K 16..23
// lane l>=16: row m=l-16, VGPRs 0-3 = K 8..15, VGPRs 4-7 = K 24..31
__device__ __forceinline__ v16bf load_frag_A(const __bf16* base, int stride, int lane) {
  int m = lane & 15, h = lane >> 4;
  const char* p = (const char*)(base + m * stride);
  FragU u;
  u.q[0] = *(const uint4*)(p + 2 * (8 * h));          // K = 8h .. 8h+7
  u.q[1] = *(const uint4*)(p + 2 * (16 + 8 * h));     // K = 16+8h .. 16+8h+7
  return u.v;
}

// B fragment (bf16 32x16) from a transposed [N][K] source:
// lane l<16 : col n=l, VGPRs 0-3 = K 0..7, VGPRs 4-7 = K 16..23 ; lane>=16: same col, K 8..15, 24..31
__device__ __forceinline__ v16bf load_frag_B(const __bf16* base, int stride, int lane) {
  int n = lane & 15, h = lane >> 4;
  const char* p = (const char*)(base + n * stride);
  FragU u;
  u.q[0] = *(const uint4*)(p + 2 * (8 * h));
  u.q[1] = *(const uint4*)(p + 2 * (16 + 8 * h));
  return u.v;
}

// pack two f32 -> one dword holding two bf16 (RNE)
__device__ __forceinline__ unsigned int pk2bf(float lo, float hi) {
  union { __bf16 b[2]; unsigned int u; } r;
  r.b[0] = (__bf16)lo; r.b[1] = (__bf16)hi;
  return r.u;
}

// ---- weight pre-transpose + bf16 conversion (W1T: [256][512], W2T: [64][256]) ----
__global__ void k_convW1(const float* __restrict__ W1, __bf16* __restrict__ W1T) {
  int idx = blockIdx.x * 256 + threadIdx.x;
  if (idx >= HID * KPAD) return;
  int n = idx >> 9, k = idx & (KPAD - 1);
  float v = (k < INCH) ? W1[k * HID + n] : 0.f;
  W1T[n * KPAD + k] = (__bf16)v;
}
__global__ void k_convW2(const float* __restrict__ W2, __bf16* __restrict__ W2T) {
  int idx = blockIdx.x * 256 + threadIdx.x;
  if (idx >= OUTCH * HID) return;
  int n = idx >> 8, k = idx & 255;
  W2T[n * HID + k] = (__bf16)W2[k * OUTCH + n];
}

// ---- fused MLP: h = relu(x@W1+b1)@W2+b2, one 64-row block per workgroup ----
__global__ __launch_bounds__(256) void k_mlp(const float* __restrict__ x,
                                             const float* __restrict__ b1,
                                             const float* __restrict__ b2,
                                             const __bf16* __restrict__ W1T,
                                             const __bf16* __restrict__ W2T,
                                             float* __restrict__ h, int nnodes) {
  __shared__ __bf16 As[2][64 * 32];  // double-buffered x tile, bf16
  __shared__ __bf16 H1[64 * HID];    // relu(layer1) tile, bf16
  const int m0 = blockIdx.x * 64;
  const int t = threadIdx.x;
  const int lane = t & 31;
  const int wave = t >> 5;

  // staging geometry (hoisted out of the k-loop)
  const int srow = t >> 2;               // 0..63
  const int scb  = (t & 3) * 8;          // 0,8,16,24
  const bool rowok = (m0 + srow) < nnodes;
  const float* xrow = x + (long long)(m0 + srow) * INCH;

  auto stage = [&](int kk, __bf16* dst) {
    uint4 pack;
    if (rowok && (kk + scb + 8 <= INCH)) {          // interior fast path
      float4 a = *(const float4*)(xrow + kk + scb);
      float4 b = *(const float4*)(xrow + kk + scb + 4);
      pack.x = pk2bf(a.x, a.y);
      pack.y = pk2bf(a.z, a.w);
      pack.z = pk2bf(b.x, b.y);
      pack.w = pk2bf(b.z, b.w);
    } else {                                        // zero-pad K>=500 / rows >= nnodes
      float v[8];
#pragma unroll
      for (int c = 0; c < 8; ++c) {
        int k = kk + scb + c;
        v[c] = (rowok && k < INCH) ? xrow[k] : 0.f;
      }
      pack.x = pk2bf(v[0], v[1]);
      pack.y = pk2bf(v[2], v[3]);
      pack.z = pk2bf(v[4], v[5]);
      pack.w = pk2bf(v[6], v[7]);
    }
    *(uint4*)(dst + srow * 32 + scb) = pack;
  };

  // -------- layer 1: 64x256 output tile; wave owns 16 rows x 128 cols --------
  const int mt  = wave & 3;          // M tile (16 rows)
  const int nt0 = (wave >> 2) * 8;   // first of 8 N tiles
  v8f zero = {};
  v8f acc[8];
#pragma unroll
  for (int j = 0; j < 8; ++j) acc[j] = zero;

  stage(0, As[0]);
  __syncthreads();
  for (int ki = 0; ki < NKT; ++ki) {
    const int kk = ki * 32;
    if (ki + 1 < NKT) stage(kk + 32, As[(ki + 1) & 1]);   // prefetch next tile
    v16bf a = load_frag_A(As[ki & 1] + mt * 16 * 32, 32, lane);
#pragma unroll
    for (int j = 0; j < 8; ++j) {
      v16bf b = load_frag_B(W1T + (nt0 + j) * 16 * KPAD + kk, KPAD, lane);
      acc[j] = __builtin_amdgcn_wmma_f32_16x16x32_bf16(false, a, false, b,
                                                       (short)0, acc[j], false, false);
    }
    __syncthreads();   // staged buffer ready / current buffer free for restage
  }
  { // epilogue: +b1, relu, bf16 -> H1
    int lm = lane & 15, lh = lane >> 4;
#pragma unroll
    for (int j = 0; j < 8; ++j) {
      int col = (nt0 + j) * 16 + lm;
      float bias = b1[col];
#pragma unroll
      for (int r = 0; r < 8; ++r) {
        int row = mt * 16 + r + 8 * lh;
        float v = acc[j][r] + bias;
        H1[row * HID + col] = (__bf16)(v > 0.f ? v : 0.f);
      }
    }
  }
  __syncthreads();

  // -------- layer 2: 64x64 output; wave owns 16 rows x 32 cols --------
  const int mt2 = wave >> 1;
  const int nb  = (wave & 1) * 2;
  v8f acc2[2];
  acc2[0] = zero; acc2[1] = zero;
  for (int kk = 0; kk < HID; kk += 32) {
    v16bf a = load_frag_A(H1 + mt2 * 16 * HID + kk, HID, lane);
#pragma unroll
    for (int j = 0; j < 2; ++j) {
      v16bf b = load_frag_B(W2T + (nb + j) * 16 * HID + kk, HID, lane);
      acc2[j] = __builtin_amdgcn_wmma_f32_16x16x32_bf16(false, a, false, b,
                                                        (short)0, acc2[j], false, false);
    }
  }
  { // epilogue: +b2, store h
    int lm = lane & 15, lh = lane >> 4;
#pragma unroll
    for (int j = 0; j < 2; ++j) {
      int col = (nb + j) * 16 + lm;
      float bias = b2[col];
#pragma unroll
      for (int r = 0; r < 8; ++r) {
        int row = mt2 * 16 + r + 8 * lh;
        int gr = m0 + row;
        if (gr < nnodes) h[(long long)gr * OUTCH + col] = acc2[j][r] + bias;
      }
    }
  }
}

// ---- graph preprocessing ----
__global__ void k_zero(float* __restrict__ p, int n) {
  int i = blockIdx.x * 256 + threadIdx.x;
  if (i < n) p[i] = 0.f;
}
__global__ void k_count(const long long* __restrict__ ei, int* __restrict__ row32,
                        int* __restrict__ col32, float* __restrict__ deg, int E) {
  int e = blockIdx.x * 256 + threadIdx.x;
  if (e >= E) return;
  int r = (int)ei[e];
  int c = (int)ei[(long long)E + e];
  row32[e] = r; col32[e] = c;
  unsafeAtomicAdd(&deg[c], 1.0f);
}
__global__ void k_dinv(float* __restrict__ deg_dinv, float* __restrict__ sfn, int n) {
  int i = blockIdx.x * 256 + threadIdx.x;
  if (i >= n) return;
  float v = rsqrtf(deg_dinv[i] + 1.0f);   // +1 self loop; always > 0
  deg_dinv[i] = v;
  sfn[i] = v * v;
}
__global__ void k_enorm(const int* __restrict__ row32, const int* __restrict__ col32,
                        const float* __restrict__ dinv, float* __restrict__ nrm, int E) {
  int e = blockIdx.x * 256 + threadIdx.x;
  if (e >= E) return;
  nrm[e] = dinv[row32[e]] * dinv[col32[e]];
}

// ---- APPNP propagation ----
__global__ void k_seed(const float* __restrict__ h, const float* __restrict__ zin,
                       const float* __restrict__ sfn, float* __restrict__ zout,
                       long long total) {
  long long i = (long long)blockIdx.x * 256 + threadIdx.x;
  if (i >= total) return;
  int n = (int)(i >> 6);
  zout[i] = ALPHA * h[i] + (1.f - ALPHA) * sfn[n] * zin[i];
}
__global__ void k_scatter(const int* __restrict__ row32, const int* __restrict__ col32,
                          const float* __restrict__ nrm, const float* __restrict__ zin,
                          float* __restrict__ zout, int E) {
  long long tid = (long long)blockIdx.x * 256 + threadIdx.x;
  int e = (int)(tid >> 4);
  if (e >= E) return;
  int c4 = ((int)tid & 15) << 2;
  int r = row32[e], c = col32[e];
  float w = (1.f - ALPHA) * nrm[e];
  float4 zv = *(const float4*)(zin + (long long)r * OUTCH + c4);
  float* dst = zout + (long long)c * OUTCH + c4;
  unsafeAtomicAdd(dst + 0, w * zv.x);
  unsafeAtomicAdd(dst + 1, w * zv.y);
  unsafeAtomicAdd(dst + 2, w * zv.z);
  unsafeAtomicAdd(dst + 3, w * zv.w);
}

// ---- in-place row log-softmax over 64 channels ----
__global__ void k_logsoftmax(float* __restrict__ z, int n) {
  int i = blockIdx.x * 256 + threadIdx.x;
  if (i >= n) return;
  float* p = z + (long long)i * OUTCH;
  float4 buf[16];
  float mx = -INFINITY;
#pragma unroll
  for (int j = 0; j < 16; ++j) {
    buf[j] = ((const float4*)p)[j];
    mx = fmaxf(mx, fmaxf(fmaxf(buf[j].x, buf[j].y), fmaxf(buf[j].z, buf[j].w)));
  }
  float s = 0.f;
#pragma unroll
  for (int j = 0; j < 16; ++j)
    s += expf(buf[j].x - mx) + expf(buf[j].y - mx) + expf(buf[j].z - mx) + expf(buf[j].w - mx);
  float lse = mx + logf(s);
#pragma unroll
  for (int j = 0; j < 16; ++j) {
    float4 o = { buf[j].x - lse, buf[j].y - lse, buf[j].z - lse, buf[j].w - lse };
    ((float4*)p)[j] = o;
  }
}

extern "C" void kernel_launch(void* const* d_in, const int* in_sizes, int n_in,
                              void* d_out, int out_size, void* d_ws, size_t ws_size,
                              hipStream_t stream) {
  const float*     x  = (const float*)d_in[0];
  const long long* ei = (const long long*)d_in[1];
  const float*     W1 = (const float*)d_in[2];
  const float*     b1 = (const float*)d_in[3];
  const float*     W2 = (const float*)d_in[4];
  const float*     b2 = (const float*)d_in[5];
  const int N = in_sizes[0] / INCH;
  const int E = in_sizes[1] / 2;

  char* w = (char*)d_ws;
  __bf16* W1T = (__bf16*)w; w += (size_t)HID * KPAD * 2;
  __bf16* W2T = (__bf16*)w; w += (size_t)OUTCH * HID * 2;
  float* h    = (float*)w;  w += (size_t)N * OUTCH * 4;
  float* zA   = (float*)w;  w += (size_t)N * OUTCH * 4;
  int*   row32= (int*)w;    w += (size_t)E * 4;
  int*   col32= (int*)w;    w += (size_t)E * 4;
  float* nrm  = (float*)w;  w += (size_t)E * 4;
  float* dinv = (float*)w;  w += (size_t)N * 4;   // deg, then d^-0.5 in place
  float* sfn  = (float*)w;  w += (size_t)N * 4;

  k_convW1<<<(HID * KPAD + 255) / 256, 256, 0, stream>>>(W1, W1T);
  k_convW2<<<(OUTCH * HID + 255) / 256, 256, 0, stream>>>(W2, W2T);
  k_zero  <<<(N + 255) / 256, 256, 0, stream>>>(dinv, N);
  k_count <<<(E + 255) / 256, 256, 0, stream>>>(ei, row32, col32, dinv, E);
  k_dinv  <<<(N + 255) / 256, 256, 0, stream>>>(dinv, sfn, N);
  k_enorm <<<(E + 255) / 256, 256, 0, stream>>>(row32, col32, dinv, nrm, E);
  k_mlp   <<<(N + 63) / 64, 256, 0, stream>>>(x, b1, b2, W1T, W2T, h, N);

  const long long total = (long long)N * OUTCH;
  const float* zin = h;
  float* zB = (float*)d_out;
  for (int it = 0; it < KPROP; ++it) {
    float* zout = (it & 1) ? zB : zA;
    k_seed<<<(int)((total + 255) / 256), 256, 0, stream>>>(h, zin, sfn, zout, total);
    k_scatter<<<(int)(((long long)E * 16 + 255) / 256), 256, 0, stream>>>(
        row32, col32, nrm, zin, zout, E);
    zin = zout;
  }
  // KPROP even -> last output is zB == d_out
  k_logsoftmax<<<(N + 255) / 256, 256, 0, stream>>>(zB, N);
}
